// MPNNModel_14482629722785
// MI455X (gfx1250) — matrix operand order repaired and checked by reading, not compiled, not run on hardware
//
#include <hip/hip_runtime.h>
#include <hip/hip_bf16.h>
#include <math.h>

// ---------------------------------------------------------------------------
// MPNN on MI455X (gfx1250, wave32).
// Heavy GEMMs run on v_wmma_f32_16x16x32_bf16: fp32 master state, bf16 operands.
// Weight B-fragments are staged in LDS per block so the vmem pipe is reserved
// for the h[src] gathers and the fp32 atomic scatter (both L2-resident).
//
// WMMA fragment layouts (CDNA5 ISA):
//   A (16x32 bf16, MxK): lane L (0..15) = row M=L;  lanes 16..31 = same rows.
//     elems 0..7  -> K = (L>=16? 8:0) + 0..7
//     elems 8..15 -> K = 16 + (L>=16? 8:0) + 0..7
//   B (32x16 bf16, KxN): lane L holds column N = L&15;
//     elems 0..7  -> K = (L>=16? 8:0) + 0..7;  elems 8..15 -> K = 16 + (L>=16? 8:0) + 0..7
//   C/D (16x16 f32): lane L holds column N = L&15; elem j -> row M = j + 8*(L>=16)
// ---------------------------------------------------------------------------

typedef __attribute__((ext_vector_type(16))) __bf16 v16bf;
typedef __attribute__((ext_vector_type(8)))  __bf16 bf16x8;
typedef __attribute__((ext_vector_type(8)))  float  v8f;
typedef __attribute__((ext_vector_type(8)))  float  f32x8;

#define DEV __device__ __forceinline__

DEV v8f wmma_bf16(v16bf a, v16bf b, v8f c) {
  return __builtin_amdgcn_wmma_f32_16x16x32_bf16(
      /*neg_a=*/false, a, /*neg_b=*/false, b,
      /*c_mod=*/(short)0, c, /*reuse_a=*/false, /*reuse_b=*/false);
}

// A-fragment from a row-major bf16 row (row has >= 32+hi8 valid elements)
DEV v16bf load_a_bf16(const __bf16* row, int hi8) {
  union { v16bf v; bf16x8 h[2]; } u;
  u.h[0] = *(const bf16x8*)(row + hi8);
  u.h[1] = *(const bf16x8*)(row + 16 + hi8);
  return u.v;
}

// A-fragment from a row-major f32 row, converting to bf16
DEV v16bf load_a_f32(const float* row, int hi8) {
  f32x8 lo = *(const f32x8*)(row + hi8);
  f32x8 hv = *(const f32x8*)(row + 16 + hi8);
  v16bf v;
#pragma unroll
  for (int e = 0; e < 8; ++e) { v[e] = (__bf16)lo[e]; v[8 + e] = (__bf16)hv[e]; }
  return v;
}

// A-fragment from an f32 row with only 16 valid K values (K 16..31 are zero pad)
DEV v16bf load_a_f32_pad16(const float* row, int hi8) {
  f32x8 lo = *(const f32x8*)(row + hi8);
  v16bf v;
#pragma unroll
  for (int e = 0; e < 8; ++e) { v[e] = (__bf16)lo[e]; v[8 + e] = (__bf16)0.0f; }
  return v;
}

// B-fragment read (32B per lane); works for both global and LDS pointers
DEV v16bf bfrag(const __bf16* packed, int lane) {
  return *(const v16bf*)(packed + lane * 16);
}

DEV void wait_lds() { asm volatile("s_wait_dscnt 0" ::: "memory"); }

// cooperative copy of nfrags weight fragments (nfrags*1024 bytes) into LDS
DEV void stage_weights(__bf16* dst, const __bf16* src, int nfrags, int tid, int nthreads) {
  const uint4* s = (const uint4*)src;
  uint4* d = (uint4*)dst;
  int total = nfrags * 64;               // 64 x 16B per fragment
  for (int i = tid; i < total; i += nthreads) d[i] = s[i];
}

// ---------------------------------------------------------------------------
// Weight packer: W [Krows x 64] row-major f32 -> bf16 B-fragments
// frag index f = ntile*kchunks + kc ; per-frag 32 lanes x 16 elems
// ---------------------------------------------------------------------------
__global__ void pack_b_kernel(const float* __restrict__ W, int Krows, int kchunks,
                              __bf16* __restrict__ out, int total) {
  int tid = blockIdx.x * blockDim.x + threadIdx.x;
  if (tid >= total) return;
  int e    = tid & 15;
  int lane = (tid >> 4) & 31;
  int f    = tid >> 9;
  int n    = f / kchunks;
  int kc   = f % kchunks;
  int koff = ((e >> 3) << 4) + ((lane >> 4) << 3) + (e & 7);  // B layout: K = 16*(e>=8) + 8*(lane>=16) + (e&7)
  int K    = kc * 32 + koff;
  int col  = n * 16 + (lane & 15);
  float v  = (K < Krows) ? W[K * 64 + col] : 0.0f;
  out[tid] = (__bf16)v;
}

__global__ void zero_kernel(float* __restrict__ p, long long n) {
  long long i = (long long)blockIdx.x * blockDim.x + threadIdx.x;
  long long stride = (long long)gridDim.x * blockDim.x;
  for (; i < n; i += stride) p[i] = 0.0f;
}

// ---------------------------------------------------------------------------
// Node encoder: h = x @ node_W + node_b   ([N,32] @ [32,64])
// ---------------------------------------------------------------------------
__global__ void __launch_bounds__(128)
node_enc_kernel(const float* __restrict__ x, const __bf16* __restrict__ pW,
                const float* __restrict__ nb, float* __restrict__ h_f32,
                __bf16* __restrict__ h_bf, int numTiles) {
  const int lane = threadIdx.x & 31;
  const int w    = threadIdx.x >> 5;
  const int tile = blockIdx.x * 4 + w;
  if (tile >= numTiles) return;
  const int lo  = lane & 15;
  const int hi8 = (lane >> 4) << 3;
  const int node = tile * 16 + lo;

  v16bf a = load_a_f32(x + (long long)node * 32, hi8);
#pragma unroll
  for (int n = 0; n < 4; ++n) {
    v8f c = {};
    c = wmma_bf16(a, bfrag(pW + n * 512, lane), c);
    float bias = nb[n * 16 + lo];
#pragma unroll
    for (int j = 0; j < 8; ++j) {
      long long nd = (long long)(tile * 16 + j + hi8) * 64 + n * 16 + lo;
      float val = c[j] + bias;
      h_f32[nd] = val;
      h_bf[nd]  = (__bf16)val;
    }
  }
}

// ---------------------------------------------------------------------------
// Message pass: per 16-edge tile (one wave each):
//   ea  = edge_attr @ We + be                       (4 WMMA, K=16 padded)
//   mid = relu(concat(h[src], ea) @ W1 + b1)        (16 WMMA)
//   m   = mid @ W2 + b2                             (8 WMMA)
//   agg[dst] += m                                   (atomic f32 scatter, L2)
// All 28 weight fragments (28 KB) live in LDS; reads go down the DS pipe.
// ---------------------------------------------------------------------------
__global__ void __launch_bounds__(128)
msg_kernel(const __bf16* __restrict__ h_bf, const float* __restrict__ edge_attr,
           const int* __restrict__ ei, const __bf16* __restrict__ pWe,
           const float* __restrict__ be, const __bf16* __restrict__ pW1,
           const float* __restrict__ b1, const __bf16* __restrict__ pW2,
           const float* __restrict__ b2, float* __restrict__ agg, int E) {
  __shared__ __align__(32) __bf16 ldsW[28 * 512];   // [We:4][W1:16][W2:8] frags
  __shared__ __align__(32) float ldsM[4][16 * 64];
  __shared__ int ldsD[4][16];

  // stage all weight fragments once per block
  stage_weights(ldsW,            pWe,  4, threadIdx.x, 128);
  stage_weights(ldsW +  4 * 512, pW1, 16, threadIdx.x, 128);
  stage_weights(ldsW + 20 * 512, pW2,  8, threadIdx.x, 128);
  __syncthreads();
  const __bf16* wE = ldsW;
  const __bf16* w1 = ldsW + 4 * 512;
  const __bf16* w2 = ldsW + 20 * 512;

  const int lane = threadIdx.x & 31;
  const int w    = threadIdx.x >> 5;
  const int tile = blockIdx.x * 4 + w;            // E/16 divisible by 4
  const int lo   = lane & 15;
  const int hi8  = (lane >> 4) << 3;
  const int m    = tile * 16 + lo;
  const int src  = ei[m];
  float* L = ldsM[w];

  // --- stage 1: edge-feature encoder tile -> LDS (cols 0..63 = ea) ---
  v16bf ae = load_a_f32_pad16(edge_attr + (long long)m * 16, hi8);
#pragma unroll
  for (int n = 0; n < 4; ++n) {
    v8f c = {};
    c = wmma_bf16(ae, bfrag(wE + n * 512, lane), c);
    float bias = be[n * 16 + lo];
#pragma unroll
    for (int j = 0; j < 8; ++j)
      L[(j + hi8) * 64 + n * 16 + lo] = c[j] + bias;
  }
  if (lane < 16) ldsD[w][lo] = ei[E + m];         // dst indices for scatter
  wait_lds();

  // --- stage 2: GEMM1 over K=128 = [h[src] | ea] ---
  v16bf a0 = load_a_bf16(h_bf + (long long)src * 64, hi8);
  v16bf a1 = load_a_bf16(h_bf + (long long)src * 64 + 32, hi8);
  v16bf a2 = load_a_f32(L + lo * 64, hi8);
  v16bf a3 = load_a_f32(L + lo * 64 + 32, hi8);
  v8f c[4];
#pragma unroll
  for (int n = 0; n < 4; ++n) {
    const __bf16* bp = w1 + n * 4 * 512;
    v8f acc = {};
    acc = wmma_bf16(a0, bfrag(bp + 0 * 512, lane), acc);
    acc = wmma_bf16(a1, bfrag(bp + 1 * 512, lane), acc);
    acc = wmma_bf16(a2, bfrag(bp + 2 * 512, lane), acc);
    acc = wmma_bf16(a3, bfrag(bp + 3 * 512, lane), acc);
    c[n] = acc;
  }

  // --- stage 3: bias + relu, D-layout -> LDS, reload as A-layout ---
#pragma unroll
  for (int n = 0; n < 4; ++n) {
    float bias = b1[n * 16 + lo];
#pragma unroll
    for (int j = 0; j < 8; ++j)
      L[(j + hi8) * 64 + n * 16 + lo] = fmaxf(c[n][j] + bias, 0.0f);
  }
  wait_lds();
  v16bf m0 = load_a_f32(L + lo * 64, hi8);
  v16bf m1 = load_a_f32(L + lo * 64 + 32, hi8);

  // --- stage 4: GEMM2 + bias + atomic scatter to agg[dst] ---
#pragma unroll
  for (int n = 0; n < 4; ++n) {
    const __bf16* bp = w2 + n * 2 * 512;
    v8f acc = {};
    acc = wmma_bf16(m0, bfrag(bp, lane), acc);
    acc = wmma_bf16(m1, bfrag(bp + 512, lane), acc);
    float bias = b2[n * 16 + lo];
#pragma unroll
    for (int j = 0; j < 8; ++j) {
      int dn = ldsD[w][j + hi8];
      atomicAdd(&agg[(long long)dn * 64 + n * 16 + lo], acc[j] + bias);
    }
  }
}

// ---------------------------------------------------------------------------
// Node update: h = relu(concat(h, agg) @ W1 + b1) @ W2 + b2   (in place)
// ---------------------------------------------------------------------------
__global__ void __launch_bounds__(128)
upd_kernel(float* __restrict__ h_f32, __bf16* __restrict__ h_bf,
           const float* __restrict__ agg, const __bf16* __restrict__ pW1,
           const float* __restrict__ b1, const __bf16* __restrict__ pW2,
           const float* __restrict__ b2, int numTiles) {
  __shared__ __align__(32) __bf16 ldsW[24 * 512];   // [W1:16][W2:8] frags
  __shared__ __align__(32) float ldsM[4][16 * 64];

  stage_weights(ldsW,            pW1, 16, threadIdx.x, 128);
  stage_weights(ldsW + 16 * 512, pW2,  8, threadIdx.x, 128);
  __syncthreads();
  const __bf16* w1 = ldsW;
  const __bf16* w2 = ldsW + 16 * 512;

  const int lane = threadIdx.x & 31;
  const int w    = threadIdx.x >> 5;
  const int tile = blockIdx.x * 4 + w;
  if (tile >= numTiles) return;
  const int lo  = lane & 15;
  const int hi8 = (lane >> 4) << 3;
  const int node = tile * 16 + lo;
  float* L = ldsM[w];

  v16bf a0 = load_a_bf16(h_bf + (long long)node * 64, hi8);
  v16bf a1 = load_a_bf16(h_bf + (long long)node * 64 + 32, hi8);
  v16bf a2 = load_a_f32(agg + (long long)node * 64, hi8);
  v16bf a3 = load_a_f32(agg + (long long)node * 64 + 32, hi8);
  v8f c[4];
#pragma unroll
  for (int n = 0; n < 4; ++n) {
    const __bf16* bp = w1 + n * 4 * 512;
    v8f acc = {};
    acc = wmma_bf16(a0, bfrag(bp + 0 * 512, lane), acc);
    acc = wmma_bf16(a1, bfrag(bp + 1 * 512, lane), acc);
    acc = wmma_bf16(a2, bfrag(bp + 2 * 512, lane), acc);
    acc = wmma_bf16(a3, bfrag(bp + 3 * 512, lane), acc);
    c[n] = acc;
  }
#pragma unroll
  for (int n = 0; n < 4; ++n) {
    float bias = b1[n * 16 + lo];
#pragma unroll
    for (int j = 0; j < 8; ++j)
      L[(j + hi8) * 64 + n * 16 + lo] = fmaxf(c[n][j] + bias, 0.0f);
  }
  wait_lds();
  v16bf m0 = load_a_f32(L + lo * 64, hi8);
  v16bf m1 = load_a_f32(L + lo * 64 + 32, hi8);
#pragma unroll
  for (int n = 0; n < 4; ++n) {
    const __bf16* bp = w2 + n * 2 * 512;
    v8f acc = {};
    acc = wmma_bf16(m0, bfrag(bp, lane), acc);
    acc = wmma_bf16(m1, bfrag(bp + 512, lane), acc);
    float bias = b2[n * 16 + lo];
#pragma unroll
    for (int j = 0; j < 8; ++j) {
      long long nd = (long long)(tile * 16 + j + hi8) * 64 + n * 16 + lo;
      float val = acc[j] + bias;
      h_f32[nd] = val;
      h_bf[nd]  = (__bf16)val;
    }
  }
}

// ---------------------------------------------------------------------------
// Global mean pool (atomic scatter) + readout MLP
// ---------------------------------------------------------------------------
__global__ void pool_kernel(const float* __restrict__ h, const int* __restrict__ batch,
                            float* __restrict__ sums, float* __restrict__ cnt,
                            long long total) {
  long long i = (long long)blockIdx.x * blockDim.x + threadIdx.x;
  long long stride = (long long)gridDim.x * blockDim.x;
  for (; i < total; i += stride) {
    int node = (int)(i >> 6);
    int c = (int)(i & 63);
    int g = batch[node];
    atomicAdd(&sums[g * 64 + c], h[i]);
    if (c == 0) atomicAdd(&cnt[g], 1.0f);
  }
}

__global__ void __launch_bounds__(128)
readout_kernel(const float* __restrict__ sums, const float* __restrict__ cnt,
               const float* __restrict__ desc, const float* __restrict__ W1,
               const float* __restrict__ b1, const float* __restrict__ W2,
               const float* __restrict__ b2, float* __restrict__ out) {
  __shared__ float r[264];
  __shared__ float red[128];
  const int g = blockIdx.x;
  const int t = threadIdx.x;
  if (t < 64) {
    float cn = fmaxf(cnt[g], 1.0f);
    r[t] = sums[g * 64 + t] / cn;
  }
  for (int i = t; i < 200; i += 128) r[64 + i] = desc[g * 200 + i];
  __syncthreads();
  float hsum = b1[t];
  for (int i = 0; i < 264; ++i) hsum += r[i] * W1[i * 128 + t];
  red[t] = fmaxf(hsum, 0.0f) * W2[t];
  __syncthreads();
  for (int s = 64; s > 0; s >>= 1) {
    if (t < s) red[t] += red[t + s];
    __syncthreads();
  }
  if (t == 0) out[g] = 1.0f / (1.0f + expf(-(red[0] + b2[0])));
}

// ---------------------------------------------------------------------------
extern "C" void kernel_launch(void* const* d_in, const int* in_sizes, int n_in,
                              void* d_out, int out_size, void* d_ws, size_t ws_size,
                              hipStream_t stream) {
  (void)in_sizes; (void)n_in; (void)out_size; (void)ws_size;
  const int N = 100000, E = 1600000, G = 1000;

  const float* x     = (const float*)d_in[0];
  const int*   ei    = (const int*)d_in[1];
  const float* eattr = (const float*)d_in[2];
  const int*   batch = (const int*)d_in[3];
  const float* desc  = (const float*)d_in[4];
  const float* nodeW = (const float*)d_in[5];
  const float* nodeb = (const float*)d_in[6];
  const float* edgeW = (const float*)d_in[7];
  const float* edgeb = (const float*)d_in[8];
  const float* emW1  = (const float*)d_in[9];
  const float* emb1  = (const float*)d_in[10];
  const float* emW2  = (const float*)d_in[11];
  const float* emb2  = (const float*)d_in[12];
  const float* umW1  = (const float*)d_in[13];
  const float* umb1  = (const float*)d_in[14];
  const float* umW2  = (const float*)d_in[15];
  const float* umb2  = (const float*)d_in[16];
  const float* roW1  = (const float*)d_in[17];
  const float* rob1  = (const float*)d_in[18];
  const float* roW2  = (const float*)d_in[19];
  const float* rob2  = (const float*)d_in[20];
  float* out = (float*)d_out;

  // workspace carve-out (~65 MB)
  char* ws = (char*)d_ws;
  size_t off = 0;
  auto carve = [&](size_t bytes) -> void* {
    void* p = ws + off;
    off = (off + bytes + 255) & ~(size_t)255;
    return p;
  };
  __bf16* p_nodeW = (__bf16*)carve(4 * 512 * 2);
  __bf16* p_edgeW = (__bf16*)carve(4 * 512 * 2);
  __bf16* p_em1   = (__bf16*)carve(3 * 16 * 512 * 2);
  __bf16* p_em2   = (__bf16*)carve(3 * 8 * 512 * 2);
  __bf16* p_um1   = (__bf16*)carve(3 * 16 * 512 * 2);
  __bf16* p_um2   = (__bf16*)carve(3 * 8 * 512 * 2);
  float*  h_f32   = (float*)carve((size_t)N * 64 * 4);
  __bf16* h_bf    = (__bf16*)carve((size_t)N * 64 * 2);
  float*  agg     = (float*)carve((size_t)N * 64 * 4);
  float*  sums    = (float*)carve((size_t)G * 64 * 4 + (size_t)G * 4);
  float*  cnt     = sums + G * 64;

  // pack all weights to bf16 B-fragments
  auto pack = [&](const float* W, int Krows, int kchunks, __bf16* dst) {
    int total = 4 * kchunks * 512;
    pack_b_kernel<<<(total + 255) / 256, 256, 0, stream>>>(W, Krows, kchunks, dst, total);
  };
  pack(nodeW, 32, 1, p_nodeW);
  pack(edgeW, 16, 1, p_edgeW);
  for (int l = 0; l < 3; ++l) {
    pack(emW1 + l * 128 * 64, 128, 4, p_em1 + l * 16 * 512);
    pack(emW2 + l * 64 * 64,  64,  2, p_em2 + l * 8 * 512);
    pack(umW1 + l * 128 * 64, 128, 4, p_um1 + l * 16 * 512);
    pack(umW2 + l * 64 * 64,  64,  2, p_um2 + l * 8 * 512);
  }

  const int nodeTiles  = N / 16;                 // 6250
  const int nodeBlocks = (nodeTiles + 3) / 4;    // 1563
  node_enc_kernel<<<nodeBlocks, 128, 0, stream>>>(x, p_nodeW, nodeb, h_f32, h_bf, nodeTiles);

  const long long hElems = (long long)N * 64;
  const int msgBlocks = E / 16 / 4;              // 25000
  for (int l = 0; l < 3; ++l) {
    zero_kernel<<<4096, 256, 0, stream>>>(agg, hElems);
    msg_kernel<<<msgBlocks, 128, 0, stream>>>(h_bf, eattr, ei, p_edgeW, edgeb,
                                              p_em1 + l * 16 * 512, emb1 + l * 64,
                                              p_em2 + l * 8 * 512, emb2 + l * 64, agg, E);
    upd_kernel<<<nodeBlocks, 128, 0, stream>>>(h_f32, h_bf, agg,
                                               p_um1 + l * 16 * 512, umb1 + l * 64,
                                               p_um2 + l * 8 * 512, umb2 + l * 64, nodeTiles);
  }

  zero_kernel<<<64, 256, 0, stream>>>(sums, (long long)G * 64 + G);
  pool_kernel<<<8192, 256, 0, stream>>>(h_f32, batch, sums, cnt, hElems);
  readout_kernel<<<G, 128, 0, stream>>>(sums, cnt, desc, roW1, rob1, roW2, rob2, out);
}
